// TopKAttentionEdgePool_25786983645322
// MI455X (gfx1250) — matrix operand order repaired and checked by reading, not compiled, not run on hardware
//
#include <hip/hip_runtime.h>
#include <hip/hip_bf16.h>
#include <math.h>

// Problem constants (from reference)
#define NB   4
#define MA   256
#define NFS  128
#define KTOP 8
#define HID  256
#define OUT  32
#define KDIM 256   // 2*NFS
#define RCUT 5.0f
#define PI_F 3.14159265358979323846f

typedef __attribute__((ext_vector_type(16))) _Float16 v16h;
typedef __attribute__((ext_vector_type(8)))  _Float16 v8h;
typedef __attribute__((ext_vector_type(8)))  float    v8f;

// ---------------------------------------------------------------------------
// K1: per-row halves of u@Wa:  a_i = h[row]·Wa[0:128],  b_j = h[row]·Wa[128:256]
// 1024 blocks (row = n*MA+i) x 128 threads
// ---------------------------------------------------------------------------
__global__ void k_rowdots(const float* __restrict__ h,
                          const float* __restrict__ Wa,
                          float* __restrict__ aW, float* __restrict__ bW) {
  int row = blockIdx.x;
  int f   = threadIdx.x;
  __shared__ float sa[128], sb[128];
  float hv = h[row * NFS + f];
  sa[f] = hv * Wa[f];
  sb[f] = hv * Wa[NFS + f];
  __syncthreads();
  for (int s = 64; s > 0; s >>= 1) {
    if (f < s) { sa[f] += sa[f + s]; sb[f] += sb[f + s]; }
    __syncthreads();
  }
  if (f == 0) { aW[row] = sa[0]; bW[row] = sb[0]; }
}

// ---------------------------------------------------------------------------
// K2: convert + swizzle weights into WMMA-fragment-major f16 layout, and pass
// z (as float) and r through to the output tuple slots.
//
// Fragment layout: half index = ((ct*8 + kk)*32 + lane)*16 + e
//   col  = ct*16 + (lane & 15)
//   krow = kk*32 + ((e>>3)<<4) + ((lane>>4)<<3) + (e&7)   (B frag: table K order)
// => each lane's 16 B-halves are 32 contiguous bytes -> 2x global_load_b128.
// ---------------------------------------------------------------------------
__global__ void k_convert(const float* __restrict__ Wp1, const float* __restrict__ Wp2,
                          const int* __restrict__ z, const float* __restrict__ r,
                          _Float16* __restrict__ Wp1s, _Float16* __restrict__ Wp2s,
                          float* __restrict__ out) {
  const int NW1 = HID * KDIM;   // 65536 swizzled halves
  const int NW2 = HID * OUT;    // 8192  swizzled halves
  const int TOT = NW1 + NW2 + NB * MA + NB * MA * 3;
  for (int t = blockIdx.x * blockDim.x + threadIdx.x; t < TOT;
       t += gridDim.x * blockDim.x) {
    int e = t;
    if (e < NW1) {
      int el   = e & 15;
      int lane = (e >> 4) & 31;
      int kk   = (e >> 9) & 7;
      int ct   = e >> 12;                      // 0..15
      int col  = (ct << 4) + (lane & 15);
      int krow = (kk << 5) + ((el >> 3) << 4) + ((lane >> 4) << 3) + (el & 7);
      Wp1s[e] = (_Float16)Wp1[krow * HID + col];
      continue;
    }
    e -= NW1;
    if (e < NW2) {
      int el   = e & 15;
      int lane = (e >> 4) & 31;
      int kk   = (e >> 9) & 7;
      int ct   = e >> 12;                      // 0..1
      int col  = (ct << 4) + (lane & 15);
      int krow = (kk << 5) + ((el >> 3) << 4) + ((lane >> 4) << 3) + (el & 7);
      Wp2s[e] = (_Float16)Wp2[krow * OUT + col];
      continue;
    }
    e -= NW2;
    if (e < NB * MA) { out[e] = (float)z[e]; continue; }   // tuple slot 0: z
    e -= NB * MA;
    out[NB * MA + e] = r[e];                                // tuple slot 1: r
  }
}

// ---------------------------------------------------------------------------
// K3: attention scores + top-K per (n,i) row.
// 1024 blocks (b = n*MA+i) x 256 threads (thread j)
// ---------------------------------------------------------------------------
__global__ void k_attn_topk(const int* __restrict__ z, const float* __restrict__ r,
                            const float* __restrict__ aW, const float* __restrict__ bW,
                            const float* __restrict__ ba,
                            float* __restrict__ attW, int* __restrict__ idxW,
                            float* __restrict__ dvkW) {
  int b = blockIdx.x;
  int n = b >> 8;
  int i = b & (MA - 1);
  int j = threadIdx.x;

  __shared__ float sg[MA];
  __shared__ float sdx[MA], sdy[MA], sdz[MA];
  __shared__ float rv[MA];
  __shared__ int   rid[MA];
  __shared__ float Ssh;

  const float* ri = r + (n * MA + i) * 3;
  const float* rj = r + (n * MA + j) * 3;
  float dx = rj[0] - ri[0];
  float dy = rj[1] - ri[1];
  float dz = rj[2] - ri[2];
  float d  = sqrtf(dx * dx + dy * dy + dz * dz);
  float cf = 0.5f * (cosf(PI_F * fminf(d, RCUT) / RCUT) + 1.0f);
  float mask = ((z[n * MA + i] > -1) && (z[n * MA + j] > -1) && (i != j)) ? 1.0f : 0.0f;
  float g = (aW[b] + cf * bW[n * MA + j] + ba[0]) * mask;
  g = mask > 0.0f ? expf(g) : 0.0f;          // exp(g) - (1 - mask)

  sg[j] = g; sdx[j] = dx; sdy[j] = dy; sdz[j] = dz;
  rv[j] = g;
  __syncthreads();
  for (int s = 128; s > 0; s >>= 1) {
    if (j < s) rv[j] += rv[j + s];
    __syncthreads();
  }
  if (j == 0) Ssh = fmaxf(rv[0], 1e-8f);
  __syncthreads();

  for (int k = 0; k < KTOP; ++k) {
    rv[j] = sg[j]; rid[j] = j;
    __syncthreads();
    for (int s = 128; s > 0; s >>= 1) {
      if (j < s) {
        float vo = rv[j + s]; int io = rid[j + s];
        if (vo > rv[j] || (vo == rv[j] && io < rid[j])) { rv[j] = vo; rid[j] = io; }
      }
      __syncthreads();
    }
    if (j == 0) {
      int bj = rid[0];
      int o  = b * KTOP + k;
      attW[o] = rv[0] / Ssh;
      idxW[o] = bj;
      float ex = sdx[bj], ey = sdy[bj], ez = sdz[bj];
      float nrm = fmaxf(sqrtf(ex * ex + ey * ey + ez * ez), 1e-4f);
      float inv = 1.0f / nrm;
      dvkW[o * 3 + 0] = ex * inv;
      dvkW[o * 3 + 1] = ey * inv;
      dvkW[o * 3 + 2] = ez * inv;
      sg[bj] = -1e30f;
    }
    __syncthreads();
  }
}

// A-fragment: two aligned 16B LDS chunks -> 2x ds_load_b128, no half shuffling.
__device__ __forceinline__ v16h load_a_frag(const _Float16* rowp, int kbase, int kh) {
  v8h lo = *(const v8h*)(rowp + kbase + kh);
  v8h hi = *(const v8h*)(rowp + kbase + kh + 16);
  return __builtin_shufflevector(lo, hi, 0, 1, 2, 3, 4, 5, 6, 7,
                                 8, 9, 10, 11, 12, 13, 14, 15);
}

// ---------------------------------------------------------------------------
// K4: fused edge-MLP via WMMA.  256 blocks x 256 threads (8 waves).
// Each block: 32 uk rows (two 16-row A tiles -> 2x B reuse).
// GEMM1 32x256x256: wave w covers columns [w*32, w*32+32), both A tiles
//   -> 4 accumulators, 32 WMMAs/wave.  SiLU -> xLDS (f16).
// GEMM2 32x256x32 on waves 0..3 (at = w>>1, ct = w&1).
// Scale by dvk, coalesced store of c.
// ---------------------------------------------------------------------------
__global__ void
__launch_bounds__(256)
k_mlp(const float* __restrict__ h,
      const float* __restrict__ attW, const int* __restrict__ idxW,
      const float* __restrict__ dvkW,
      const _Float16* __restrict__ Wp1s, const _Float16* __restrict__ Wp2s,
      const float* __restrict__ bp1, const float* __restrict__ bp2,
      float* __restrict__ outc) {
  __shared__ _Float16 ukLDS[32][KDIM];   // 16 KB
  __shared__ _Float16 xLDS[32][HID];     // 16 KB
  __shared__ float    c2LDS[32][OUT];    // 4 KB

  const int t       = threadIdx.x;
  const int wave    = t >> 5;
  const int lane    = t & 31;
  const int rowbase = blockIdx.x * 32;   // of 8192 uk rows

  // keep swizzled Wp1 hot in L2 (global_prefetch_b8)
  __builtin_prefetch(Wp1s + (size_t)t * 256, 0, 0);

  // ---- build uk tile: uk[row] = concat(h[n,i,:], h[n,idx,:]) * att ----
#pragma unroll 4
  for (int rl = 0; rl < 32; ++rl) {
    int row = rowbase + rl;
    int n   = row >> 11;            // /(MA*KTOP)
    int i   = (row >> 3) & (MA - 1);
    float att = attW[row];
    int   jx  = idxW[row];
    float v = (t < NFS) ? h[(n * MA + i) * NFS + t]
                        : h[(n * MA + jx) * NFS + (t - NFS)];
    ukLDS[rl][t] = (_Float16)(v * att);
  }
  __syncthreads();

  // ---- GEMM1 ----
  const int m   = lane & 15;
  const int kh  = (lane >> 4) << 3;     // A-frag K sub-offset: 0 or 8
  const int cn  = lane & 15;            // C-frag column within tile
  const int mo  = (lane >> 4) << 3;     // C-frag row sub-offset: 0 or 8
  const int ct0 = wave * 2;
  const int ct1 = wave * 2 + 1;
  const int cb0 = ct0 * 16;
  const int cb1 = ct1 * 16;

  const v16h* B1 = (const v16h*)Wp1s;   // fragment-major: [(ct*8+kk)*32 + lane]

  v8f acc00 = {}; v8f acc01 = {}; v8f acc10 = {}; v8f acc11 = {};
#pragma unroll
  for (int kk = 0; kk < 8; ++kk) {
    const int kbase = kk * 32;
    v16h a0 = load_a_frag(&ukLDS[m][0],      kbase, kh);
    v16h a1 = load_a_frag(&ukLDS[16 + m][0], kbase, kh);
    v16h b0 = B1[(ct0 * 8 + kk) * 32 + lane];   // 2x global_load_b128
    v16h b1 = B1[(ct1 * 8 + kk) * 32 + lane];
    acc00 = __builtin_amdgcn_wmma_f32_16x16x32_f16(false, a0, false, b0,
                                                   (short)0, acc00, false, false);
    acc01 = __builtin_amdgcn_wmma_f32_16x16x32_f16(false, a0, false, b1,
                                                   (short)0, acc01, false, false);
    acc10 = __builtin_amdgcn_wmma_f32_16x16x32_f16(false, a1, false, b0,
                                                   (short)0, acc10, false, false);
    acc11 = __builtin_amdgcn_wmma_f32_16x16x32_f16(false, a1, false, b1,
                                                   (short)0, acc11, false, false);
  }

  // ---- bias + SiLU -> xLDS (f16) ----
  const float bia0 = bp1[cb0 + cn];
  const float bia1 = bp1[cb1 + cn];
#pragma unroll
  for (int p = 0; p < 8; ++p) {
    int mr0 = p + mo;
    int mr1 = 16 + p + mo;
    float x00 = acc00[p] + bia0;
    float x01 = acc01[p] + bia1;
    float x10 = acc10[p] + bia0;
    float x11 = acc11[p] + bia1;
    xLDS[mr0][cb0 + cn] = (_Float16)(x00 / (1.0f + expf(-x00)));
    xLDS[mr0][cb1 + cn] = (_Float16)(x01 / (1.0f + expf(-x01)));
    xLDS[mr1][cb0 + cn] = (_Float16)(x10 / (1.0f + expf(-x10)));
    xLDS[mr1][cb1 + cn] = (_Float16)(x11 / (1.0f + expf(-x11)));
  }
  __syncthreads();

  // ---- GEMM2: (32x256) x (256x32) on waves 0..3 ----
  if (wave < 4) {
    const int at  = wave >> 1;          // A tile 0/1
    const int ct2 = wave & 1;           // column tile 0/1
    const int cb  = ct2 * 16;
    const v16h* B2 = (const v16h*)Wp2s;
    v8f acc2 = {};
#pragma unroll
    for (int kk = 0; kk < 8; ++kk) {
      const int kbase = kk * 32;
      v16h a  = load_a_frag(&xLDS[at * 16 + m][0], kbase, kh);
      v16h b2 = B2[(ct2 * 8 + kk) * 32 + lane];
      acc2 = __builtin_amdgcn_wmma_f32_16x16x32_f16(false, a, false, b2,
                                                    (short)0, acc2, false, false);
    }
    const float bia2 = bp2[cb + cn];
#pragma unroll
    for (int p = 0; p < 8; ++p)
      c2LDS[at * 16 + p + mo][cb + cn] = acc2[p] + bia2;
  }
  __syncthreads();

  // ---- c[row, o, d] = C2[row, o] * dvk[row, d], coalesced store ----
  for (int e = t; e < 32 * OUT * 3; e += 256) {
    int rl  = e / (OUT * 3);
    int rem = e - rl * (OUT * 3);
    int o   = rem / 3;
    int d   = rem - o * 3;
    int row = rowbase + rl;
    outc[(size_t)(row * OUT + o) * 3 + d] = c2LDS[rl][o] * dvkW[row * 3 + d];
  }
}

// ---------------------------------------------------------------------------
extern "C" void kernel_launch(void* const* d_in, const int* in_sizes, int n_in,
                              void* d_out, int out_size, void* d_ws, size_t ws_size,
                              hipStream_t stream) {
  const int*   z   = (const int*)  d_in[0];
  const float* r   = (const float*)d_in[1];
  const float* h   = (const float*)d_in[2];
  const float* Wa  = (const float*)d_in[3];
  const float* ba  = (const float*)d_in[4];
  const float* Wp1 = (const float*)d_in[5];
  const float* bp1 = (const float*)d_in[6];
  const float* Wp2 = (const float*)d_in[7];
  const float* bp2 = (const float*)d_in[8];
  float* out = (float*)d_out;

  // workspace layout (bytes)
  char* ws = (char*)d_ws;
  float*    aW   = (float*)(ws + 0);            // 1024 f
  float*    bW   = (float*)(ws + 4096);         // 1024 f
  float*    attW = (float*)(ws + 8192);         // 8192 f
  float*    dvkW = (float*)(ws + 40960);        // 24576 f
  int*      idxW = (int*)  (ws + 139264);       // 8192 i
  _Float16* Wp1s = (_Float16*)(ws + 172032);    // 65536 h (fragment-major)
  _Float16* Wp2s = (_Float16*)(ws + 303104);    // 8192 h  (fragment-major)
  // total ~312 KB

  k_rowdots<<<dim3(NB * MA), dim3(NFS), 0, stream>>>(h, Wa, aW, bW);

  const int tot2 = HID * KDIM + HID * OUT + NB * MA + NB * MA * 3;
  k_convert<<<dim3((tot2 + 255) / 256), dim3(256), 0, stream>>>(
      Wp1, Wp2, z, r, Wp1s, Wp2s, out);

  k_attn_topk<<<dim3(NB * MA), dim3(MA), 0, stream>>>(
      z, r, aW, bW, ba, attW, idxW, dvkW);

  float* outc = out + NB * MA + NB * MA * 3;   // after z and r tuple slots
  k_mlp<<<dim3(NB * MA * KTOP / 32), dim3(256), 0, stream>>>(
      h, attW, idxW, dvkW, Wp1s, Wp2s, bp1, bp2, outc);
}
